// NeuralLongTermMemory_33689723470202
// MI455X (gfx1250) — matrix operand order repaired and checked by reading, not compiled, not run on hardware
//
#include <hip/hip_runtime.h>
#include <stdint.h>

#define D_  512
#define H_  1024
#define B_  4
#define S_  4096
#define NC_ 64          // number of chunks
#define BC_ 256         // tokens per chunk (B * CHUNK)
#define HD_ 524288      // H_*D_

typedef __bf16 v16bf __attribute__((ext_vector_type(16)));
typedef float  v8f   __attribute__((ext_vector_type(8)));
typedef unsigned short u16;
typedef unsigned int   u32;

union V16U { v16bf v; uint4 q[2]; };

__device__ __forceinline__ float bf2f(u16 u) {
  union { u32 u; float f; } c; c.u = ((u32)u) << 16; return c.f;
}
__device__ __forceinline__ u16 f2bf(float f) {
  union { float f; u32 u; } c; c.f = f;
  u32 r = c.u + 0x7FFFu + ((c.u >> 16) & 1u);
  return (u16)(r >> 16);
}
__device__ __forceinline__ float sigm(float x) { return 1.0f / (1.0f + __expf(-x)); }
__device__ __forceinline__ float clip1(float x) { return fminf(1.0f, fmaxf(-1.0f, x)); }

// ---- WMMA fragment loads (bf16, K-contiguous storage) -----------------------
// A: [M][K] row-major.  lane m = lane&15; K pattern 16g + 8h + (0..7)
__device__ __forceinline__ v16bf ldA(const u16* A, int ldk, int m0, int k0, int lane) {
  const int m  = m0 + (lane & 15);
  const int kb = k0 + 8 * (lane >> 4);
  const u16* p = A + (size_t)m * ldk + kb;
  V16U u;
  u.q[0] = *(const uint4*)(p);        // K = kb .. kb+7
  u.q[1] = *(const uint4*)(p + 16);   // K = kb+16 .. kb+23
  return u.v;
}
// B: stored transposed as [N][K] row-major. lane n = lane&15; K pattern 16g + 8h + (0..7)
__device__ __forceinline__ v16bf ldB(const u16* Bt, int ldk, int n0, int k0, int lane) {
  const int n  = n0 + (lane & 15);
  const int kb = k0 + 8 * (lane >> 4);
  const u16* p = Bt + (size_t)n * ldk + kb;
  V16U u;
  u.q[0] = *(const uint4*)(p);
  u.q[1] = *(const uint4*)(p + 16);
  return u.v;
}

// 32x32 output tile per wave: 2x2 WMMA tiles with A/B fragment reuse.
__device__ __forceinline__ void gemm32x32(const u16* A, const u16* B, int K,
                                          int m0, int n0, int lane, v8f acc[2][2]) {
  for (int k = 0; k < K; k += 32) {
    v16bf a0 = ldA(A, K, m0,      k, lane);
    v16bf a1 = ldA(A, K, m0 + 16, k, lane);
    v16bf b0 = ldB(B, K, n0,      k, lane);
    v16bf b1 = ldB(B, K, n0 + 16, k, lane);
    acc[0][0] = __builtin_amdgcn_wmma_f32_16x16x32_bf16(false, a0, false, b0, (short)0, acc[0][0], false, false);
    acc[0][1] = __builtin_amdgcn_wmma_f32_16x16x32_bf16(false, a0, false, b1, (short)0, acc[0][1], false, false);
    acc[1][0] = __builtin_amdgcn_wmma_f32_16x16x32_bf16(false, a1, false, b0, (short)0, acc[1][0], false, false);
    acc[1][1] = __builtin_amdgcn_wmma_f32_16x16x32_bf16(false, a1, false, b1, (short)0, acc[1][1], false, false);
  }
}

#define TILE_SETUP \
  const int lane = threadIdx.x; const int wave = threadIdx.y; \
  const int m0 = blockIdx.y * 64  + (wave >> 2) * 32; \
  const int n0 = blockIdx.x * 128 + (wave & 3)  * 32; \
  v8f acc[2][2] = {};

#define EPILOGUE_BEGIN \
  const int rb = 8 * (lane >> 4); const int nc = lane & 15; \
  for (int tm = 0; tm < 2; ++tm) for (int tn = 0; tn < 2; ++tn) { \
    v8f cfr = acc[tm][tn]; \
    for (int r = 0; r < 8; ++r) { \
      const int m = m0 + tm * 16 + rb + r; \
      const int n = n0 + tn * 16 + nc; \
      float val = cfr[r];

#define EPILOGUE_END } }

// ---- utility kernels --------------------------------------------------------
__global__ void k_zero(float* p, int n) {
  int i = blockIdx.x * 256 + threadIdx.x;
  if (i < n) p[i] = 0.0f;
}
__global__ void k_f2b(const float* s, u16* d, int n) {
  int i = blockIdx.x * 256 + threadIdx.x;
  if (i < n) d[i] = f2bf(s[i]);
}

// projections: [16384 x 512] @ W^T, write bf16 into chunk-major [nc][BC][D]
__global__ __launch_bounds__(256) void k_proj(const u16* A, const u16* W, u16* dst) {
  TILE_SETUP;
  gemm32x32(A, W, 512, m0, n0, lane, acc);
  EPILOGUE_BEGIN
    int b = m >> 12, s = m & 4095;
    int ch = s >> 6, c = s & 63;
    int t = b * 64 + c;
    dst[((size_t)(ch * BC_ + t)) * D_ + n] = f2bf(val);
  EPILOGUE_END
}

// row-wise L2 normalize (in place, bf16); optional transposed scatter [nc][D][BC]
__global__ void k_norm(u16* tsr, u16* tT, int doT) {
  __shared__ float red[256];
  const int row = blockIdx.x, tid = threadIdx.x;
  u16* rp = tsr + (size_t)row * D_;
  float v0 = bf2f(rp[tid]), v1 = bf2f(rp[tid + 256]);
  red[tid] = v0 * v0 + v1 * v1;
  __syncthreads();
  for (int s2 = 128; s2 > 0; s2 >>= 1) {
    if (tid < s2) red[tid] += red[tid + s2];
    __syncthreads();
  }
  float inv = 1.0f / (sqrtf(red[0]) + 1e-8f);
  float n0 = v0 * inv, n1 = v1 * inv;
  rp[tid] = f2bf(n0); rp[tid + 256] = f2bf(n1);
  if (doT) {
    int ch = row >> 8, t = row & 255;
    u16* base = tT + (size_t)ch * D_ * BC_;
    base[(size_t)tid * BC_ + t]         = f2bf(n0);
    base[(size_t)(tid + 256) * BC_ + t] = f2bf(n1);
  }
}

// per-chunk scalar gates: sigmoid(x.w + b) averaged over batch & chunk tokens
__global__ void k_gates(const float* x, const float* gdw, const float* gdb,
                        const float* glw, const float* glb,
                        const float* gmw, const float* gmb,
                        float* dec, float* lr, float* eta) {
  __shared__ float s0[256], s1[256], s2[256];
  const int ch = blockIdx.x, tid = threadIdx.x;
  const int b = tid >> 6, c = tid & 63, s = ch * 64 + c;
  const float* xr = x + ((size_t)b * S_ + s) * D_;
  float a0 = 0.f, a1 = 0.f, a2 = 0.f;
  for (int d = 0; d < D_; ++d) {
    float xv = xr[d];
    a0 = fmaf(xv, gdw[d], a0);
    a1 = fmaf(xv, glw[d], a1);
    a2 = fmaf(xv, gmw[d], a2);
  }
  s0[tid] = sigm(a0 + gdb[0]);
  s1[tid] = sigm(a1 + glb[0]);
  s2[tid] = sigm(a2 + gmb[0]);
  __syncthreads();
  for (int st = 128; st > 0; st >>= 1) {
    if (tid < st) { s0[tid] += s0[tid + st]; s1[tid] += s1[tid + st]; s2[tid] += s2[tid + st]; }
    __syncthreads();
  }
  if (tid == 0) {
    dec[ch] = s0[0] * (1.0f / 256.0f);
    lr[ch]  = s1[0] * (1.0f / 256.0f);
    eta[ch] = s2[0] * (1.0f / 256.0f);
  }
}

// effective weights (bf16) for the inner step: e1=Wm1+d1 [H][D]; e2/e2T=Wm2+d2
__global__ void k_prep(const float* Wm1, const float* d1,
                       const float* Wm2, const float* d2,
                       u16* e1, u16* e2, u16* e2T) {
  int i = blockIdx.x * 256 + threadIdx.x;    // i = h*D_ + d
  int h = i >> 9, d = i & 511;
  e1[i] = f2bf(Wm1[i] + d1[i]);
  int j = d * H_ + h;
  u16 wb = f2bf(Wm2[j] + d2[j]);
  e2[j] = wb;
  e2T[i] = wb;
}

// G1: pre1 = kt @ e1^T  (M=256,N=1024,K=512). a1 = silu(pre1), store a1 and a1^T.
__global__ __launch_bounds__(256) void k_g1(const u16* kt, const u16* e1,
                                            float* pre1, u16* a1o, u16* a1T) {
  TILE_SETUP;
  gemm32x32(kt, e1, 512, m0, n0, lane, acc);
  EPILOGUE_BEGIN
    pre1[m * H_ + n] = val;
    float sv = val * sigm(val);
    u16 sb = f2bf(sv);
    a1o[m * H_ + n] = sb;
    a1T[n * BC_ + m] = sb;
  EPILOGUE_END
}

// G2: pred = a1 @ e2^T (M=256,N=512,K=1024). err clip, db2 scale, store db2 & db2^T.
__global__ __launch_bounds__(256) void k_g2(const u16* a1, const u16* e2,
                                            const u16* vch, u16* db2, u16* db2T) {
  TILE_SETUP;
  gemm32x32(a1, e2, 1024, m0, n0, lane, acc);
  const float scale = 2.0f / 131072.0f;    // 2 / (B*CHUNK*D)
  EPILOGUE_BEGIN
    float vv = bf2f(vch[m * D_ + n]);
    float err = clip1(val - vv);
    u16 db = f2bf(scale * err);
    db2[m * D_ + n] = db;
    db2T[n * BC_ + m] = db;
  EPILOGUE_END
}

// G3: g2 = db2^T @ a1 (M=512(d),N=1024(h),K=256). Update m2,d2; r2 = Wm2 + d2n.
__global__ __launch_bounds__(256) void k_g3(const u16* db2T, const u16* a1T,
                                            float* m2, float* d2, const float* Wm2,
                                            u16* r2, const float* decv, const float* lrv,
                                            const float* etav, int ch) {
  TILE_SETUP;
  gemm32x32(db2T, a1T, 256, m0, n0, lane, acc);
  const float dec = decv[ch], lr = lrv[ch], eta = etav[ch];
  EPILOGUE_BEGIN
    float g2 = clip1(val);
    int idx = m * H_ + n;
    float mm = eta * m2[idx] - lr * g2;
    m2[idx] = mm;
    float dd = (1.0f - dec) * d2[idx] + mm;
    d2[idx] = dd;
    r2[idx] = f2bf(Wm2[idx] + dd);
  EPILOGUE_END
}

// G4: db1 = (db2 @ e2) * silu'(pre1)  (M=256,N=1024,K=512). Store db1^T.
__global__ __launch_bounds__(256) void k_g4(const u16* db2, const u16* e2T,
                                            const float* pre1, u16* db1T) {
  TILE_SETUP;
  gemm32x32(db2, e2T, 512, m0, n0, lane, acc);
  EPILOGUE_BEGIN
    float p = pre1[m * H_ + n];
    float sg = sigm(p);
    float ds = sg * (1.0f + p * (1.0f - sg));
    db1T[n * BC_ + m] = f2bf(val * ds);
  EPILOGUE_END
}

// G5: g1 = db1^T @ kt (M=1024(h),N=512(d),K=256). Update m1,d1; r1 = Wm1 + d1n.
__global__ __launch_bounds__(256) void k_g5(const u16* db1T, const u16* ktT,
                                            float* m1, float* d1, const float* Wm1,
                                            u16* r1, const float* decv, const float* lrv,
                                            const float* etav, int ch) {
  TILE_SETUP;
  gemm32x32(db1T, ktT, 256, m0, n0, lane, acc);
  const float dec = decv[ch], lr = lrv[ch], eta = etav[ch];
  EPILOGUE_BEGIN
    float g1 = clip1(val);
    int idx = m * D_ + n;
    float mm = eta * m1[idx] - lr * g1;
    m1[idx] = mm;
    float dd = (1.0f - dec) * d1[idx] + mm;
    d1[idx] = dd;
    r1[idx] = f2bf(Wm1[idx] + dd);
  EPILOGUE_END
}

// G6: h = silu(qt @ r1^T) (M=256,N=1024,K=512)
__global__ __launch_bounds__(256) void k_g6(const u16* qt, const u16* r1, u16* hh) {
  TILE_SETUP;
  gemm32x32(qt, r1, 512, m0, n0, lane, acc);
  EPILOGUE_BEGIN
    hh[m * H_ + n] = f2bf(val * sigm(val));
  EPILOGUE_END
}

// G7: y = h @ r2^T (M=256,N=512,K=1024), write bf16 chunk-major
__global__ __launch_bounds__(256) void k_g7(const u16* hh, const u16* r2, u16* ych) {
  TILE_SETUP;
  gemm32x32(hh, r2, 1024, m0, n0, lane, acc);
  EPILOGUE_BEGIN
    ych[m * D_ + n] = f2bf(val);
  EPILOGUE_END
}

// G8: out = y @ Wout^T (M=16384,N=512,K=512); map chunk-major rows -> [B,S,D] fp32
__global__ __launch_bounds__(256) void k_g8(const u16* ybf, const u16* wo, float* out) {
  TILE_SETUP;
  gemm32x32(ybf, wo, 512, m0, n0, lane, acc);
  EPILOGUE_BEGIN
    int ch = m >> 8, t = m & 255;
    int b = t >> 6, c = t & 63;
    int s = ch * 64 + c;
    out[((size_t)(b * S_ + s)) * D_ + n] = val;
  EPILOGUE_END
}

// ---------------------------------------------------------------------------
extern "C" void kernel_launch(void* const* d_in, const int* in_sizes, int n_in,
                              void* d_out, int out_size, void* d_ws, size_t ws_size,
                              hipStream_t stream) {
  const float* x    = (const float*)d_in[0];
  const float* Wk   = (const float*)d_in[1];
  const float* Wv   = (const float*)d_in[2];
  const float* Wq   = (const float*)d_in[3];
  const float* Wout = (const float*)d_in[4];
  const float* Wm1  = (const float*)d_in[5];
  const float* Wm2  = (const float*)d_in[6];
  const float* gdw  = (const float*)d_in[7];
  const float* gdb  = (const float*)d_in[8];
  const float* glw  = (const float*)d_in[9];
  const float* glb  = (const float*)d_in[10];
  const float* gmw  = (const float*)d_in[11];
  const float* gmb  = (const float*)d_in[12];
  float* out = (float*)d_out;

  char* w = (char*)d_ws;
  size_t o = 0;
  auto alloc = [&](size_t bytes) -> char* {
    char* r = w + o;
    o = (o + bytes + 255) & ~(size_t)255;
    return r;
  };

  // fp32 state (d1,m1 contiguous with d2,m2 so one zero pass covers all)
  float* d1  = (float*)alloc(HD_ * 4);
  float* m1  = (float*)alloc(HD_ * 4);
  float* d2  = (float*)alloc(HD_ * 4);
  float* m2  = (float*)alloc(HD_ * 4);
  float* dec = (float*)alloc(NC_ * 4);
  float* lr  = (float*)alloc(NC_ * 4);
  float* eta = (float*)alloc(NC_ * 4);
  float* pre1 = (float*)alloc((size_t)BC_ * H_ * 4);
  // bf16 buffers
  u16* xbf  = (u16*)alloc((size_t)B_ * S_ * D_ * 2);
  u16* wkb  = (u16*)alloc((size_t)D_ * D_ * 2);
  u16* wvb  = (u16*)alloc((size_t)D_ * D_ * 2);
  u16* wqb  = (u16*)alloc((size_t)D_ * D_ * 2);
  u16* wob  = (u16*)alloc((size_t)D_ * D_ * 2);
  u16* kc   = (u16*)alloc((size_t)NC_ * BC_ * D_ * 2);
  u16* qc   = (u16*)alloc((size_t)NC_ * BC_ * D_ * 2);
  u16* vc   = (u16*)alloc((size_t)NC_ * BC_ * D_ * 2);
  u16* ybf  = (u16*)alloc((size_t)NC_ * BC_ * D_ * 2);
  u16* kcT  = (u16*)alloc((size_t)NC_ * D_ * BC_ * 2);
  u16* e1   = (u16*)alloc((size_t)HD_ * 2);
  u16* e2   = (u16*)alloc((size_t)HD_ * 2);
  u16* e2T  = (u16*)alloc((size_t)HD_ * 2);
  u16* r1   = (u16*)alloc((size_t)HD_ * 2);
  u16* r2   = (u16*)alloc((size_t)HD_ * 2);
  u16* a1   = (u16*)alloc((size_t)BC_ * H_ * 2);
  u16* a1T  = (u16*)alloc((size_t)H_ * BC_ * 2);
  u16* db2  = (u16*)alloc((size_t)BC_ * D_ * 2);
  u16* db2T = (u16*)alloc((size_t)D_ * BC_ * 2);
  u16* db1T = (u16*)alloc((size_t)H_ * BC_ * 2);
  u16* hh   = (u16*)alloc((size_t)BC_ * H_ * 2);
  (void)ws_size; (void)n_in; (void)in_sizes; (void)out_size;

  const dim3 blk(32, 8, 1);
  // grid = (N/128, M/64)
  const dim3 gProj(4, 256), gMN_256_1024(8, 4), gMN_256_512(4, 4);
  const dim3 gMN_512_1024(8, 8), gMN_1024_512(4, 16), gOut(4, 256);

  // init fast-weight state to zero (4 contiguous HD_ buffers)
  k_zero<<<(4 * HD_) / 256, 256, 0, stream>>>(d1, 4 * HD_);

  // bf16 conversions
  k_f2b<<<(B_ * S_ * D_) / 256, 256, 0, stream>>>(x, xbf, B_ * S_ * D_);
  k_f2b<<<(D_ * D_) / 256, 256, 0, stream>>>(Wk, wkb, D_ * D_);
  k_f2b<<<(D_ * D_) / 256, 256, 0, stream>>>(Wv, wvb, D_ * D_);
  k_f2b<<<(D_ * D_) / 256, 256, 0, stream>>>(Wq, wqb, D_ * D_);
  k_f2b<<<(D_ * D_) / 256, 256, 0, stream>>>(Wout, wob, D_ * D_);

  // projections (WMMA) into chunk-major bf16
  k_proj<<<gProj, blk, 0, stream>>>(xbf, wkb, kc);
  k_proj<<<gProj, blk, 0, stream>>>(xbf, wvb, vc);
  k_proj<<<gProj, blk, 0, stream>>>(xbf, wqb, qc);

  // l2 normalize k (with transpose) and q
  k_norm<<<NC_ * BC_, 256, 0, stream>>>(kc, kcT, 1);
  k_norm<<<NC_ * BC_, 256, 0, stream>>>(qc, (u16*)nullptr, 0);

  // per-chunk gates
  k_gates<<<NC_, 256, 0, stream>>>(x, gdw, gdb, glw, glb, gmw, gmb, dec, lr, eta);

  // sequential scan over chunks
  for (int ch = 0; ch < NC_; ++ch) {
    const size_t coff  = (size_t)ch * BC_ * D_;   // chunk-major [BC][D]
    const size_t coffT = (size_t)ch * D_ * BC_;   // chunk-major [D][BC]
    k_prep<<<HD_ / 256, 256, 0, stream>>>(Wm1, d1, Wm2, d2, e1, e2, e2T);
    k_g1<<<gMN_256_1024, blk, 0, stream>>>(kc + coff, e1, pre1, a1, a1T);
    k_g2<<<gMN_256_512, blk, 0, stream>>>(a1, e2, vc + coff, db2, db2T);
    k_g3<<<gMN_512_1024, blk, 0, stream>>>(db2T, a1T, m2, d2, Wm2, r2, dec, lr, eta, ch);
    k_g4<<<gMN_256_1024, blk, 0, stream>>>(db2, e2T, pre1, db1T);
    k_g5<<<gMN_1024_512, blk, 0, stream>>>(db1T, kcT + coffT, m1, d1, Wm1, r1, dec, lr, eta, ch);
    k_g6<<<gMN_256_1024, blk, 0, stream>>>(qc + coff, r1, hh);
    k_g7<<<gMN_256_512, blk, 0, stream>>>(hh, r2, ybf + coff);
  }

  // final output projection
  k_g8<<<gOut, blk, 0, stream>>>(ybf, wob, out);
}
